// GAN_Module_2989297238600
// MI455X (gfx1250) — compile-verified, matrix operand order checked by reading, not run on hardware
//
#include <hip/hip_runtime.h>
#include <hip/hip_bf16.h>
#include <cstdint>

#define B_   2
#define N_   16384
#define E_   131072
#define C_   256
#define H_   8
#define CIN_ 259
#define KPAD_ 288     // 9 chunks of 32
#define SAP_  296     // LDS pitch for 288-wide A panel (16B aligned rows)
#define SAPU_ 264     // LDS pitch for 256-wide A panel (k_update)

typedef __attribute__((ext_vector_type(16))) _Float16 v16h;
typedef __attribute__((ext_vector_type(8)))  float    v8f;
typedef __attribute__((ext_vector_type(8)))  _Float16 h8;
typedef __attribute__((ext_vector_type(2)))  _Float16 h2;

// ---------------- WMMA fragment helpers ----------------
// A fragment for chunk at rowBase = sAll + (lane&15)*pitch + kc*32:
// elements 0..7  = rowBase[half*8 .. half*8+7]        (ds_load_b128)
// elements 8..15 = rowBase[16+half*8 .. 16+half*8+7]  (ds_load_b128)
__device__ inline v16h ldA2(const _Float16* p, int half) {
  h8 lo = *(const h8*)(p + half * 8);
  h8 hi = *(const h8*)(p + 16 + half * 8);
  return __builtin_shufflevector(lo, hi, 0, 1, 2, 3, 4, 5, 6, 7,
                                 8, 9, 10, 11, 12, 13, 14, 15);
}
// B fragment from pre-packed weights: lane's 16 f16 contiguous (32B) ->
// two global_load_b128, zero shuffle VALU.
__device__ inline v16h ldBp(const _Float16* __restrict__ Wp, int kc, int ntile, int lane) {
  return *(const v16h*)(Wp + ((((size_t)kc * (C_ / 16) + ntile) * 32 + lane) * 16));
}
__device__ inline v8f wmma_f16(v16h a, v16h b, v8f c) {
  return __builtin_amdgcn_wmma_f32_16x16x32_f16(false, a, false, b, (short)0, c, false, false);
}

// ---------------- LayerNorm stats over 16 rows x 256 cols in LDS ----------------
__device__ inline void ln_stats(const float* buf, int ld, float* part, float* part2,
                                float* mu, float* istd, int tid) {
  int m = tid >> 4, c = tid & 15;
  float s = 0.f, s2 = 0.f;
#pragma unroll
  for (int j = 0; j < 16; ++j) {
    float x = buf[m * ld + c * 16 + j];
    s += x; s2 += x * x;
  }
  part[m * 16 + c] = s;
  part2[m * 16 + c] = s2;
  __syncthreads();
  if (tid < 16) {
    float S = 0.f, S2 = 0.f;
    for (int j = 0; j < 16; ++j) { S += part[tid * 16 + j]; S2 += part2[tid * 16 + j]; }
    float mean = S * (1.f / 256.f);
    float var  = fmaxf(S2 * (1.f / 256.f) - mean * mean, 0.f);
    mu[tid]   = mean;
    istd[tid] = rsqrtf(var + 1e-5f);
  }
  __syncthreads();
}

__device__ inline void atomicMaxF(float* addr, float v) {
  if (v >= 0.f) atomicMax((int*)addr, __float_as_int(v));
  else          atomicMin((unsigned int*)addr, __float_as_uint(v));
}

// ---------------- pre-kernels ----------------
__global__ void k_featcvt(const float* __restrict__ features, _Float16* __restrict__ feat_h) {
  size_t idx = (size_t)blockIdx.x * 256 + threadIdx.x;
  if (idx >= (size_t)B_ * N_ * C_) return;
  int c = (int)(idx % C_);
  int n = (int)((idx / C_) % N_);
  int b = (int)(idx / ((size_t)C_ * N_));
  feat_h[idx] = (_Float16)features[((size_t)b * C_ + c) * N_ + n];
}

// Pack f32 weight (rows x C_, row-major) into WMMA B-fragment order.
__global__ void k_wpack(const float* __restrict__ src, _Float16* __restrict__ dst,
                        int rows, int nchunks) {
  size_t idx = (size_t)blockIdx.x * 256 + threadIdx.x;
  if (idx >= (size_t)nchunks * 8192) return;
  int elem  = (int)(idx & 15);
  int lane  = (int)((idx >> 4) & 31);
  int ntile = (int)((idx >> 9) & 15);      // C_/16 == 16
  int kc    = (int)(idx >> 13);            // 8192 == 2^13
  int half = lane >> 4, l16 = lane & 15;
  int j = elem >> 1, p = elem & 1;
  int kb = (j >> 2) * 16 + half * 8 + (j & 3) * 2 + p;
  int k = kc * 32 + kb;
  int n = ntile * 16 + l16;
  dst[idx] = (k < rows) ? (_Float16)src[(size_t)k * C_ + n] : (_Float16)0.f;
}

__global__ void k_init(float* __restrict__ agg, float* __restrict__ segmax,
                       float* __restrict__ segsum, float* __restrict__ cnt) {
  size_t idx = (size_t)blockIdx.x * 256 + threadIdx.x;
  if (idx < (size_t)B_ * N_ * C_) agg[idx] = 0.f;
  if (idx < (size_t)B_ * N_ * H_) { segmax[idx] = __int_as_float(0xff800000); segsum[idx] = 0.f; }
  if (idx < (size_t)B_ * N_) cnt[idx] = 0.f;
}

// ---------------- fused edge kernel: K,V,Q GEMMs + LN + simi ----------------
__global__ __launch_bounds__(256) void k_edge(
    const float* __restrict__ xyz, const int* __restrict__ edges,
    const _Float16* __restrict__ feat_h,
    const _Float16* __restrict__ WkP, const float* __restrict__ biask,
    const float* __restrict__ gk, const float* __restrict__ betak,
    const _Float16* __restrict__ WvP, const float* __restrict__ biasv,
    const float* __restrict__ gv, const float* __restrict__ betav,
    const _Float16* __restrict__ WqP, const float* __restrict__ biasq,
    const float* __restrict__ gq, const float* __restrict__ betaq,
    _Float16* __restrict__ Vh, float* __restrict__ simi) {
  const int b    = blockIdx.y;
  const int e0   = blockIdx.x * 16;
  const int tid  = threadIdx.x;
  const int lane = tid & 31;
  const int wave = tid >> 5;
  const int half = lane >> 4;
  const int l16  = lane & 15;
  const size_t bN = (size_t)b * N_;

  __shared__ int   s_src[16], s_dst[16];
  __shared__ float s_rel[16][3];
  __shared__ _Float16 sAll[16 * SAP_];   // full 16 x 288 A panel (f16)
  __shared__ float sK[16][260];
  __shared__ float sQ[16][260];          // also used for V / Q staging
  __shared__ float s_part[256], s_part2[256];
  __shared__ float s_mu[2][16], s_is[2][16];

  if (tid < 16) {
    size_t eidx = ((size_t)b * E_ + e0 + tid) * 2;
    int s = edges[eidx + 0];
    int d = edges[eidx + 1];
    s_src[tid] = s; s_dst[tid] = d;
#pragma unroll
    for (int j = 0; j < 3; ++j)
      s_rel[tid][j] = xyz[(bN + s) * 3 + j] - xyz[(bN + d) * 3 + j];
  }
  __syncthreads();

  // per-thread staging assignment (hoisted out of the k-loop entirely)
  const int m16 = tid >> 4;      // row this thread stages
  const int sg  = tid & 15;      // 32B segment within the row
  const _Float16* rowS = feat_h + (bN + (size_t)s_src[m16]) * C_;
  const _Float16* rowD = feat_h + (bN + (size_t)s_dst[m16]) * C_;
  h2 relp;
  {
    int k = sg * 2;
    relp.x = (_Float16)((k     < 3) ? s_rel[m16][k]     : 0.f);
    relp.y = (_Float16)((k + 1 < 3) ? s_rel[m16][k + 1] : 0.f);
  }

  const int nt0 = wave * 2;       // first 16-col tile index
  const int n0  = wave * 32;

  // ---- phase S: stage full src panel once ----
  *(h8*)(sAll + m16 * SAP_ + sg * 16)     = *(const h8*)(rowS + sg * 16);
  *(h8*)(sAll + m16 * SAP_ + sg * 16 + 8) = *(const h8*)(rowS + sg * 16 + 8);
  *(h2*)(sAll + m16 * SAP_ + 256 + sg * 2) = relp;
  __syncthreads();

  v8f aK0 = {}, aK1 = {}, aV0 = {}, aV1 = {};
  const _Float16* aRow = sAll + l16 * SAP_;
#pragma unroll
  for (int kc = 0; kc < 9; ++kc) {
    v16h a   = ldA2(aRow + kc * 32, half);
    v16h bk0 = ldBp(WkP, kc, nt0,     lane);
    v16h bk1 = ldBp(WkP, kc, nt0 + 1, lane);
    v16h bv0 = ldBp(WvP, kc, nt0,     lane);
    v16h bv1 = ldBp(WvP, kc, nt0 + 1, lane);
    aK0 = wmma_f16(a, bk0, aK0);
    aK1 = wmma_f16(a, bk1, aK1);
    aV0 = wmma_f16(a, bv0, aV0);
    aV1 = wmma_f16(a, bv1, aV1);
  }
#pragma unroll
  for (int r = 0; r < 8; ++r) {
    int m = r + half * 8;
    int n = n0 + l16;
    sK[m][n]      = fmaxf(aK0[r] + biask[n],      0.f);
    sK[m][n + 16] = fmaxf(aK1[r] + biask[n + 16], 0.f);
    sQ[m][n]      = fmaxf(aV0[r] + biasv[n],      0.f);
    sQ[m][n + 16] = fmaxf(aV1[r] + biasv[n + 16], 0.f);
  }
  __syncthreads();
  ln_stats(&sK[0][0], 260, s_part, s_part2, s_mu[0], s_is[0], tid);
  ln_stats(&sQ[0][0], 260, s_part, s_part2, s_mu[1], s_is[1], tid);
  {
    int m = tid >> 4, cb = (tid & 15) * 16;
    float muK = s_mu[0][m], isK = s_is[0][m];
    float muV = s_mu[1][m], isV = s_is[1][m];
    size_t vbase = ((size_t)b * E_ + e0 + m) * C_ + cb;
#pragma unroll
    for (int j = 0; j < 16; ++j) {
      int c = cb + j;
      sK[m][c] = (sK[m][c] - muK) * isK * gk[c] + betak[c];   // LN'd K kept in LDS
      float vx = (sQ[m][c] - muV) * isV * gv[c] + betav[c];   // LN'd V -> global f16
      Vh[vbase + j] = (_Float16)vx;
    }
  }
  __syncthreads();   // everyone done reading sAll (and sQ for V) before restaging

  // ---- phase D: stage full dst panel once ----
  *(h8*)(sAll + m16 * SAP_ + sg * 16)     = *(const h8*)(rowD + sg * 16);
  *(h8*)(sAll + m16 * SAP_ + sg * 16 + 8) = *(const h8*)(rowD + sg * 16 + 8);
  *(h2*)(sAll + m16 * SAP_ + 256 + sg * 2) = relp;
  __syncthreads();

  v8f aQ0 = {}, aQ1 = {};
#pragma unroll
  for (int kc = 0; kc < 9; ++kc) {
    v16h a   = ldA2(aRow + kc * 32, half);
    v16h bq0 = ldBp(WqP, kc, nt0,     lane);
    v16h bq1 = ldBp(WqP, kc, nt0 + 1, lane);
    aQ0 = wmma_f16(a, bq0, aQ0);
    aQ1 = wmma_f16(a, bq1, aQ1);
  }
#pragma unroll
  for (int r = 0; r < 8; ++r) {
    int m = r + half * 8;
    int n = n0 + l16;
    sQ[m][n]      = fmaxf(aQ0[r] + biasq[n],      0.f);
    sQ[m][n + 16] = fmaxf(aQ1[r] + biasq[n + 16], 0.f);
  }
  __syncthreads();
  ln_stats(&sQ[0][0], 260, s_part, s_part2, s_mu[1], s_is[1], tid);
  {
    int m = tid >> 4, cb = (tid & 15) * 16;
    float mu = s_mu[1][m], is = s_is[1][m];
#pragma unroll
    for (int j = 0; j < 16; ++j) {
      int c = cb + j;
      sQ[m][c] = (sQ[m][c] - mu) * is * gq[c] + betaq[c];
    }
  }
  __syncthreads();

  // ---- simi = (K . Q) per head / sqrt(DH) ----
  if (tid < 128) {
    int m = tid >> 3, h = tid & 7;
    float acc = 0.f;
#pragma unroll
    for (int j = 0; j < 32; ++j) acc += sK[m][h * 32 + j] * sQ[m][h * 32 + j];
    simi[((size_t)b * E_ + e0 + m) * H_ + h] = acc * 0.17677669529663687f;
  }
}

// ---------------- segment softmax ----------------
__global__ void k_segmax(const int* __restrict__ edges, const float* __restrict__ simi,
                         float* __restrict__ segmax, float* __restrict__ cnt) {
  int b = blockIdx.y;
  size_t idx = (size_t)blockIdx.x * 256 + threadIdx.x;
  if (idx >= (size_t)E_ * H_) return;
  int e = (int)(idx >> 3), h = (int)(idx & 7);
  int d = edges[((size_t)b * E_ + e) * 2 + 1];
  float v = simi[((size_t)b * E_ + e) * H_ + h];
  atomicMaxF(&segmax[((size_t)b * N_ + d) * H_ + h], v);
  if (h == 0) atomicAdd(&cnt[(size_t)b * N_ + d], 1.0f);
}

__global__ void k_expsum(const int* __restrict__ edges, float* __restrict__ simi,
                         const float* __restrict__ segmax, float* __restrict__ segsum) {
  int b = blockIdx.y;
  size_t idx = (size_t)blockIdx.x * 256 + threadIdx.x;
  if (idx >= (size_t)E_ * H_) return;
  int e = (int)(idx >> 3), h = (int)(idx & 7);
  int d = edges[((size_t)b * E_ + e) * 2 + 1];
  float mx = segmax[((size_t)b * N_ + d) * H_ + h];
  float w = __expf(simi[((size_t)b * E_ + e) * H_ + h] - mx);
  simi[((size_t)b * E_ + e) * H_ + h] = w;   // overwrite with expw
  atomicAdd(&segsum[((size_t)b * N_ + d) * H_ + h], w);
}

__global__ void k_scatter(const int* __restrict__ edges, const float* __restrict__ expw,
                          const float* __restrict__ segsum, const _Float16* __restrict__ Vh,
                          float* __restrict__ agg) {
  int b = blockIdx.y;
  size_t idx = (size_t)blockIdx.x * 256 + threadIdx.x;
  int e = (int)(idx >> 8), c = (int)(idx & 255);
  int d = edges[((size_t)b * E_ + e) * 2 + 1];
  int h = c >> 5;
  float w    = expw[((size_t)b * E_ + e) * H_ + h];
  float base = segsum[((size_t)b * N_ + d) * H_ + h];
  float v = (float)Vh[((size_t)b * E_ + e) * C_ + c];
  atomicAdd(&agg[((size_t)b * N_ + d) * C_ + c], v * (w / base));
}

__global__ void k_div(const float* __restrict__ agg, const float* __restrict__ cnt,
                      _Float16* __restrict__ aggh) {
  int b = blockIdx.y;
  size_t idx = (size_t)blockIdx.x * 256 + threadIdx.x;
  int n = (int)(idx >> 8);
  float cc = fmaxf(cnt[(size_t)b * N_ + n], 1.0f);
  size_t o = (size_t)b * N_ * C_ + idx;
  aggh[o] = (_Float16)(agg[o] / cc);
}

// ---------------- update GEMM + LN + residual ----------------
__global__ __launch_bounds__(256) void k_update(
    const _Float16* __restrict__ aggh, const _Float16* __restrict__ WuP,
    const float* __restrict__ biasu, const float* __restrict__ gu,
    const float* __restrict__ betau, const float* __restrict__ features,
    float* __restrict__ out) {
  const int b     = blockIdx.y;
  const int nrow0 = blockIdx.x * 16;   // node tile
  const int tid   = threadIdx.x;
  const int lane  = tid & 31;
  const int wave  = tid >> 5;
  const int half  = lane >> 4;
  const int l16   = lane & 15;

  __shared__ _Float16 sAll[16 * SAPU_];
  __shared__ float sO[16][260];
  __shared__ float s_part[256], s_part2[256];
  __shared__ float s_mu[16], s_is[16];

  // stage full 16 x 256 panel once
  {
    int m = tid >> 4, sg = tid & 15;
    const _Float16* row = aggh + ((size_t)b * N_ + nrow0 + m) * C_;
    *(h8*)(sAll + m * SAPU_ + sg * 16)     = *(const h8*)(row + sg * 16);
    *(h8*)(sAll + m * SAPU_ + sg * 16 + 8) = *(const h8*)(row + sg * 16 + 8);
  }
  __syncthreads();

  const int nt0 = wave * 2;
  const int n0  = wave * 32;
  const _Float16* aRow = sAll + l16 * SAPU_;
  v8f a0 = {}, a1 = {};
#pragma unroll
  for (int kc = 0; kc < 8; ++kc) {
    v16h a  = ldA2(aRow + kc * 32, half);
    v16h b0 = ldBp(WuP, kc, nt0,     lane);
    v16h b1 = ldBp(WuP, kc, nt0 + 1, lane);
    a0 = wmma_f16(a, b0, a0);
    a1 = wmma_f16(a, b1, a1);
  }
#pragma unroll
  for (int r = 0; r < 8; ++r) {
    int m = r + half * 8;
    int n = n0 + l16;
    sO[m][n]      = fmaxf(a0[r] + biasu[n],      0.f);
    sO[m][n + 16] = fmaxf(a1[r] + biasu[n + 16], 0.f);
  }
  __syncthreads();
  ln_stats(&sO[0][0], 260, s_part, s_part2, s_mu, s_is, tid);
  {
    int m = tid >> 4, cb = (tid & 15) * 16;
    float mu = s_mu[m], is = s_is[m];
#pragma unroll
    for (int j = 0; j < 16; ++j) {
      int c = cb + j;
      float val = (sO[m][c] - mu) * is * gu[c] + betau[c];
      size_t oi = ((size_t)b * C_ + c) * N_ + nrow0 + m;
      out[oi] = val + features[oi];
    }
  }
}

// ---------------- launch ----------------
extern "C" void kernel_launch(void* const* d_in, const int* in_sizes, int n_in,
                              void* d_out, int out_size, void* d_ws, size_t ws_size,
                              hipStream_t stream) {
  const float* xyz      = (const float*)d_in[0];
  const float* features = (const float*)d_in[1];
  const int*   edges    = (const int*)d_in[2];
  const float* Wk = (const float*)d_in[3];  const float* bk = (const float*)d_in[4];
  const float* gk = (const float*)d_in[5];  const float* betak = (const float*)d_in[6];
  const float* Wv = (const float*)d_in[7];  const float* bv = (const float*)d_in[8];
  const float* gv = (const float*)d_in[9];  const float* betav = (const float*)d_in[10];
  const float* Wq = (const float*)d_in[11]; const float* bq = (const float*)d_in[12];
  const float* gq = (const float*)d_in[13]; const float* betaq = (const float*)d_in[14];
  const float* Wu = (const float*)d_in[15]; const float* bu = (const float*)d_in[16];
  const float* gu = (const float*)d_in[17]; const float* betau = (const float*)d_in[18];
  float* out = (float*)d_out;

  char* ws = (char*)d_ws;
  size_t off = 0;
  auto carve = [&](size_t bytes) -> char* {
    char* p = ws + off;
    off = (off + bytes + 255) & ~(size_t)255;
    return p;
  };
  _Float16* feat_h = (_Float16*)carve((size_t)B_ * N_ * C_ * 2);
  _Float16* WkP    = (_Float16*)carve((size_t)KPAD_ * C_ * 2);
  _Float16* WvP    = (_Float16*)carve((size_t)KPAD_ * C_ * 2);
  _Float16* WqP    = (_Float16*)carve((size_t)KPAD_ * C_ * 2);
  _Float16* WuP    = (_Float16*)carve((size_t)C_ * C_ * 2);
  _Float16* Vh     = (_Float16*)carve((size_t)B_ * E_ * C_ * 2);
  float*    simi   = (float*)carve((size_t)B_ * E_ * H_ * 4);
  float*    segmax = (float*)carve((size_t)B_ * N_ * H_ * 4);
  float*    segsum = (float*)carve((size_t)B_ * N_ * H_ * 4);
  float*    cnt    = (float*)carve((size_t)B_ * N_ * 4);
  float*    agg    = (float*)carve((size_t)B_ * N_ * C_ * 4);
  _Float16* aggh   = (_Float16*)carve((size_t)B_ * N_ * C_ * 2);
  (void)ws_size; (void)in_sizes; (void)n_in; (void)out_size;

  // pre-kernels
  k_featcvt<<<(unsigned)(((size_t)B_ * N_ * C_) / 256), 256, 0, stream>>>(features, feat_h);
  k_wpack<<<(9 * 8192) / 256, 256, 0, stream>>>(Wk, WkP, CIN_, 9);
  k_wpack<<<(9 * 8192) / 256, 256, 0, stream>>>(Wv, WvP, CIN_, 9);
  k_wpack<<<(9 * 8192) / 256, 256, 0, stream>>>(Wq, WqP, CIN_, 9);
  k_wpack<<<(8 * 8192) / 256, 256, 0, stream>>>(Wu, WuP, C_, 8);
  k_init<<<(unsigned)(((size_t)B_ * N_ * C_) / 256), 256, 0, stream>>>(agg, segmax, segsum, cnt);

  // fused edge MLPs (WMMA) + simi
  k_edge<<<dim3(E_ / 16, B_), 256, 0, stream>>>(
      xyz, edges, feat_h,
      WkP, bk, gk, betak,
      WvP, bv, gv, betav,
      WqP, bq, gq, betaq,
      Vh, simi);

  // segment softmax + aggregation
  k_segmax<<<dim3((E_ * H_) / 256, B_), 256, 0, stream>>>(edges, simi, segmax, cnt);
  k_expsum<<<dim3((E_ * H_) / 256, B_), 256, 0, stream>>>(edges, simi, segmax, segsum);
  k_scatter<<<dim3((unsigned)(((size_t)E_ * C_) / 256), B_), 256, 0, stream>>>(
      edges, simi, segsum, Vh, agg);
  k_div<<<dim3((N_ * C_) / 256, B_), 256, 0, stream>>>(agg, cnt, aggh);

  // update MLP (WMMA) + LN + residual, transposed output
  k_update<<<dim3(N_ / 16, B_), 256, 0, stream>>>(aggh, WuP, bu, gu, betau, features, out);
}
